// BaseGraphEncoder_86569360818240
// MI455X (gfx1250) — matrix operand order repaired and checked by reading, not compiled, not run on hardware
//
#include <hip/hip_runtime.h>
#include <hip/hip_bf16.h>

// ---------------------------------------------------------------------------
// Types for CDNA5 WMMA
// ---------------------------------------------------------------------------
typedef __bf16 bf16_t;
typedef bf16_t v16bf __attribute__((ext_vector_type(16)));
typedef float  v8f   __attribute__((ext_vector_type(8)));
typedef unsigned short us4 __attribute__((ext_vector_type(4)));
typedef unsigned short us8 __attribute__((ext_vector_type(8)));

union Frag16 { v16bf v; us8 h[2]; };

__device__ __forceinline__ unsigned short f2bf_rne(float f) {
    unsigned int u = __float_as_uint(f);
    unsigned int r = u + 0x7FFFu + ((u >> 16) & 1u);   // round-to-nearest-even
    return (unsigned short)(r >> 16);
}

// ---------------------------------------------------------------------------
// Weight conversion: f32 (K x N) -> bf16 TRANSPOSED + padded (Npad x Kpad).
// Transposed storage makes the GEMM B-tile copy linear (async-to-LDS friendly)
// and the per-lane fragment reads contiguous b128s.
// ---------------------------------------------------------------------------
__global__ __launch_bounds__(256) void conv_w_kernel(
    const float* __restrict__ W, int din, int dout,
    unsigned short* __restrict__ outT, int dinp, int doutp)
{
    int idx = blockIdx.x * 256 + threadIdx.x;
    int n = dinp * doutp;
    if (idx >= n) return;
    int col = idx / dinp;     // output-feature (N) index
    int k   = idx % dinp;     // input-feature  (K) index
    float v = (k < din && col < dout) ? W[k * dout + col] : 0.0f;
    outT[idx] = f2bf_rne(v);
}

__global__ __launch_bounds__(256) void pad_bias_kernel(
    const float* __restrict__ b, int dact, float* __restrict__ out, int dpad)
{
    int idx = blockIdx.x * 256 + threadIdx.x;
    if (idx >= dpad) return;
    out[idx] = (idx < dact) ? b[idx] : 0.0f;
}

// ---------------------------------------------------------------------------
// float4 copy (m = h) and zero init
// ---------------------------------------------------------------------------
__global__ __launch_bounds__(256) void copy4_kernel(
    float* __restrict__ dst, const float* __restrict__ src, long long n4)
{
    long long idx = (long long)blockIdx.x * 256 + threadIdx.x;
    if (idx >= n4) return;
    ((float4*)dst)[idx] = ((const float4*)src)[idx];
}

__global__ __launch_bounds__(256) void zero_kernel(float* __restrict__ p, long long n)
{
    long long idx = (long long)blockIdx.x * 256 + threadIdx.x;
    if (idx < n) p[idx] = 0.0f;
}

// ---------------------------------------------------------------------------
// GIN aggregation: m[dst] += h[src]   (atomic f32, float4 gather granularity)
// ---------------------------------------------------------------------------
__global__ __launch_bounds__(256) void scatter_add_kernel(
    const int* __restrict__ ei, const float* __restrict__ h,
    float* __restrict__ m, int E, int W)
{
    long long t = (long long)blockIdx.x * 256 + threadIdx.x;
    int chunks = W >> 2;
    if (t >= (long long)E * chunks) return;
    int e = (int)(t / chunks);
    int c = (int)(t % chunks) * 4;
    int s = ei[e];
    int d = ei[E + e];
    float4 v = *(const float4*)&h[(long long)s * W + c];
    float* base = &m[(long long)d * W + c];
    atomicAdd(base + 0, v.x);
    atomicAdd(base + 1, v.y);
    atomicAdd(base + 2, v.z);
    atomicAdd(base + 3, v.w);
}

// ---------------------------------------------------------------------------
// bf16 WMMA GEMM (double-buffered, 2x2 fragment blocked):
//   C[M x Npad] = act( A[M x K](f32 -> bf16) * Wt[Npad x K](bf16, transposed) + bias )
// Block tile 128x64, 256 threads = 8 wave32; each wave computes 32x32
// (2 A-frags x 2 B-frags = 4 WMMAs per K-step).
// A tile: float4 global loads, RNE convert, us4 LDS stores.
// B tile: linear copy via global_load_async_to_lds_b128 (ASYNCcnt).
// Ping-pong LDS buffers -> one barrier per K-step, loads overlap compute.
// ---------------------------------------------------------------------------
#define TM 128
#define TN 64
#define TK 32
#define LDSTRIDE 40   // ushorts; 80B row stride, multiple of 16B

__global__ __launch_bounds__(256) void wmma_gemm_bias_act(
    const float* __restrict__ A, const unsigned short* __restrict__ Wt,
    const float* __restrict__ bias, float* __restrict__ C,
    int M, int K, int Npad, int Nact, int relu)
{
    __shared__ __align__(16) unsigned short As[2][TM][LDSTRIDE];
    __shared__ __align__(16) unsigned short BsT[2][TN][LDSTRIDE];

    const int tid  = threadIdx.x;
    const int lane = tid & 31;
    const int wave = tid >> 5;
    const int wm   = wave & 3;    // 4 row groups of 32
    const int wn   = wave >> 2;   // 2 col groups of 32
    const int mBase = blockIdx.y * TM;
    const int nBase = blockIdx.x * TN;

    // B-tile copy assignment: thread -> (column, 16B chunk); 64*32 ushorts = 256 chunks
    const int bcol = tid >> 2;        // 0..63
    const int bq   = (tid & 3) * 8;   // ushort offset 0,8,16,24
    const unsigned short* gB = Wt + (long long)(nBase + bcol) * K + bq;
    const unsigned int ldsB[2] = {
        (unsigned int)(uintptr_t)&BsT[0][bcol][bq],
        (unsigned int)(uintptr_t)&BsT[1][bcol][bq]
    };

    // A-tile copy assignment: thread -> (row, float4 chunk), 4 chunks/thread
    const int ar  = tid >> 3;         // 0..31 (+32*i)
    const int ac4 = (tid & 7) * 4;    // float offset 0..28

    auto loadTile = [&](int k0, int buf) {
        // async B: one 16B global->LDS copy per thread (ASYNCcnt)
        unsigned long long ga = (unsigned long long)(uintptr_t)(gB + k0);
        asm volatile("global_load_async_to_lds_b128 %0, %1, off"
                     :: "v"(ldsB[buf]), "v"(ga) : "memory");
        // A: float4 load -> bf16 RNE -> us4 LDS store
#pragma unroll
        for (int i = 0; i < 4; ++i) {
            int r = ar + i * 32;
            int row = mBase + r;
            float4 v = (row < M) ? *(const float4*)&A[(long long)row * K + k0 + ac4]
                                 : make_float4(0.f, 0.f, 0.f, 0.f);
            us4 o; o[0] = f2bf_rne(v.x); o[1] = f2bf_rne(v.y);
                   o[2] = f2bf_rne(v.z); o[3] = f2bf_rne(v.w);
            *(us4*)&As[buf][r][ac4] = o;
        }
    };

    v8f acc00 = {}, acc01 = {}, acc10 = {}, acc11 = {};

    const int kTiles = K / TK;
    loadTile(0, 0);
    int cur = 0;

    for (int t = 0; t < kTiles; ++t) {
        // Only buf[cur]'s async batch is outstanding here.
        asm volatile("s_wait_asynccnt 0x0" ::: "memory");
        __syncthreads();   // emits s_wait_dscnt 0 + barrier: buf[cur] fully visible

        if (t + 1 < kTiles) loadTile((t + 1) * TK, cur ^ 1);

        // ---- Fragments from buf[cur]: aligned 128-bit LDS reads ----
        const int hi = lane >> 4;
        const unsigned short (*Asb)[LDSTRIDE] = As[cur];
        const unsigned short (*Bsb)[LDSTRIDE] = BsT[cur];

        Frag16 a0, a1, b0, b1;
        {
            const unsigned short* ap0 = &Asb[wm * 32 + (lane & 15)][0];
            a0.h[0] = *(const us8*)(ap0 + hi * 8);
            a0.h[1] = *(const us8*)(ap0 + 16 + hi * 8);
            const unsigned short* ap1 = ap0 + 16 * LDSTRIDE;
            a1.h[0] = *(const us8*)(ap1 + hi * 8);
            a1.h[1] = *(const us8*)(ap1 + 16 + hi * 8);
        }
        {
            const unsigned short* bp0 = &Bsb[wn * 32 + (lane & 15)][hi * 8];
            b0.h[0] = *(const us8*)(bp0);
            b0.h[1] = *(const us8*)(bp0 + 16);
            const unsigned short* bp1 = bp0 + 16 * LDSTRIDE;
            b1.h[0] = *(const us8*)(bp1);
            b1.h[1] = *(const us8*)(bp1 + 16);
        }

        acc00 = __builtin_amdgcn_wmma_f32_16x16x32_bf16(
            false, a0.v, false, b0.v, (short)0, acc00, false, false);
        acc01 = __builtin_amdgcn_wmma_f32_16x16x32_bf16(
            false, a0.v, false, b1.v, (short)0, acc01, false, false);
        acc10 = __builtin_amdgcn_wmma_f32_16x16x32_bf16(
            false, a1.v, false, b0.v, (short)0, acc10, false, false);
        acc11 = __builtin_amdgcn_wmma_f32_16x16x32_bf16(
            false, a1.v, false, b1.v, (short)0, acc11, false, false);

        cur ^= 1;
    }

    // Epilogue: C/D layout — lane n = lane&15, M = (lane>>4)*8 + v
    const int nl = lane & 15;
    const int hv = lane >> 4;
#pragma unroll
    for (int mi = 0; mi < 2; ++mi) {
        v8f am0 = (mi == 0) ? acc00 : acc10;
        v8f am1 = (mi == 0) ? acc01 : acc11;
#pragma unroll
        for (int v = 0; v < 8; ++v) {
            int row = mBase + wm * 32 + mi * 16 + hv * 8 + v;
            if (row < M) {
                int c0 = nBase + wn * 32 + nl;
                int c1 = c0 + 16;
                float x0 = am0[v], x1 = am1[v];
                if (c0 < Nact) { x0 += bias[c0]; if (relu) x0 = fmaxf(x0, 0.0f); } else x0 = 0.0f;
                if (c1 < Nact) { x1 += bias[c1]; if (relu) x1 = fmaxf(x1, 0.0f); } else x1 = 0.0f;
                C[(long long)row * Npad + c0] = x0;
                C[(long long)row * Npad + c1] = x1;
            }
        }
    }
}

// ---------------------------------------------------------------------------
// Global mean pool: sums[g] += h[i], counts[g] += 1
// ---------------------------------------------------------------------------
__global__ __launch_bounds__(256) void pool_sum_kernel(
    const float* __restrict__ h, const int* __restrict__ batch,
    float* __restrict__ sums, int N)
{
    long long t = (long long)blockIdx.x * 256 + threadIdx.x;
    const int chunks = 768 / 4;
    if (t >= (long long)N * chunks) return;
    int node = (int)(t / chunks);
    int c    = (int)(t % chunks) * 4;
    int g    = batch[node];
    float4 v = *(const float4*)&h[(long long)node * 768 + c];
    float* base = &sums[(long long)g * 768 + c];
    atomicAdd(base + 0, v.x);
    atomicAdd(base + 1, v.y);
    atomicAdd(base + 2, v.z);
    atomicAdd(base + 3, v.w);
}

__global__ __launch_bounds__(256) void count_kernel(
    const int* __restrict__ batch, float* __restrict__ counts, int N)
{
    int t = blockIdx.x * 256 + threadIdx.x;
    if (t < N) atomicAdd(&counts[batch[t]], 1.0f);
}

// ---------------------------------------------------------------------------
// Per-graph LayerNorm over 768 features; one block (8 wave32) per graph
// ---------------------------------------------------------------------------
__global__ __launch_bounds__(256) void ln_kernel(
    const float* __restrict__ sums, const float* __restrict__ counts,
    const float* __restrict__ lnw, const float* __restrict__ lnb,
    float* __restrict__ out)
{
    int g = blockIdx.x;
    int t = threadIdx.x;
    float inv = 1.0f / fmaxf(counts[g], 1.0f);
    float p0 = sums[g * 768 + t]       * inv;
    float p1 = sums[g * 768 + t + 256] * inv;
    float p2 = sums[g * 768 + t + 512] * inv;

    __shared__ float red[8];
    __shared__ float stat[2];

    float s = p0 + p1 + p2;
#pragma unroll
    for (int o = 16; o > 0; o >>= 1) s += __shfl_down(s, o, 32);
    if ((t & 31) == 0) red[t >> 5] = s;
    __syncthreads();
    if (t == 0) {
        float tot = 0.0f;
        for (int i = 0; i < 8; ++i) tot += red[i];
        stat[0] = tot * (1.0f / 768.0f);
    }
    __syncthreads();
    float mu = stat[0];
    float d0 = p0 - mu, d1 = p1 - mu, d2 = p2 - mu;
    float vs = d0 * d0 + d1 * d1 + d2 * d2;
#pragma unroll
    for (int o = 16; o > 0; o >>= 1) vs += __shfl_down(vs, o, 32);
    if ((t & 31) == 0) red[t >> 5] = vs;
    __syncthreads();
    if (t == 0) {
        float tot = 0.0f;
        for (int i = 0; i < 8; ++i) tot += red[i];
        stat[1] = tot * (1.0f / 768.0f);
    }
    __syncthreads();
    float rstd = rsqrtf(stat[1] + 1e-5f);
    out[g * 768 + t]       = d0 * rstd * lnw[t]       + lnb[t];
    out[g * 768 + t + 256] = d1 * rstd * lnw[t + 256] + lnb[t + 256];
    out[g * 768 + t + 512] = d2 * rstd * lnw[t + 512] + lnb[t + 512];
}

// ---------------------------------------------------------------------------
// Host-side orchestration
// ---------------------------------------------------------------------------
extern "C" void kernel_launch(void* const* d_in, const int* in_sizes, int n_in,
                              void* d_out, int out_size, void* d_ws, size_t ws_size,
                              hipStream_t stream)
{
    const float* x     = (const float*)d_in[0];
    const int*   ei    = (const int*)d_in[1];
    const int*   batch = (const int*)d_in[2];
    const float* W1p[5]; const float* B1p[5]; const float* W2p[5]; const float* B2p[5];
    for (int i = 0; i < 5; ++i) {
        W1p[i] = (const float*)d_in[3 + 4 * i];
        B1p[i] = (const float*)d_in[4 + 4 * i];
        W2p[i] = (const float*)d_in[5 + 4 * i];
        B2p[i] = (const float*)d_in[6 + 4 * i];
    }
    const float* lnw = (const float*)d_in[23];
    const float* lnb = (const float*)d_in[24];

    const int NN = in_sizes[0] / 128;   // 50000 nodes
    const int EE = in_sizes[1] / 2;     // 800000 edges

    static const int dims[6] = {128, 300, 300, 300, 300, 768};
    static const int padd[6] = {128, 320, 320, 320, 320, 768};

    // Workspace carving
    char* base = (char*)d_ws;
    size_t off = 0;
    auto take = [&](size_t bytes) -> char* {
        size_t a = (off + 255) & ~(size_t)255;
        off = a + bytes;
        return base + a;
    };

    size_t woff[5][2], boff[5][2];
    size_t wtot = 0, btot = 0;
    for (int i = 0; i < 5; ++i) {
        woff[i][0] = wtot; wtot += (size_t)padd[i] * padd[i + 1];
        woff[i][1] = wtot; wtot += (size_t)padd[i + 1] * padd[i + 1];
        boff[i][0] = btot; btot += padd[i + 1];
        boff[i][1] = btot; btot += padd[i + 1];
    }

    float* H  = (float*)take((size_t)NN * 768 * sizeof(float));
    float* Mb = (float*)take((size_t)NN * 768 * sizeof(float));
    float* T  = (float*)take((size_t)NN * 768 * sizeof(float));
    unsigned short* Wbf = (unsigned short*)take(wtot * sizeof(unsigned short));
    float* Bf   = (float*)take(btot * sizeof(float));
    float* sums = (float*)take((size_t)256 * 768 * sizeof(float) + 256 * sizeof(float));
    float* counts = sums + 256 * 768;

    // Pre-convert weights (transposed + padded bf16) and biases (padded f32)
    for (int i = 0; i < 5; ++i) {
        int n1 = padd[i] * padd[i + 1];
        conv_w_kernel<<<(n1 + 255) / 256, 256, 0, stream>>>(
            W1p[i], dims[i], dims[i + 1], Wbf + woff[i][0], padd[i], padd[i + 1]);
        int n2 = padd[i + 1] * padd[i + 1];
        conv_w_kernel<<<(n2 + 255) / 256, 256, 0, stream>>>(
            W2p[i], dims[i + 1], dims[i + 1], Wbf + woff[i][1], padd[i + 1], padd[i + 1]);
        pad_bias_kernel<<<(padd[i + 1] + 255) / 256, 256, 0, stream>>>(
            B1p[i], dims[i + 1], Bf + boff[i][0], padd[i + 1]);
        pad_bias_kernel<<<(padd[i + 1] + 255) / 256, 256, 0, stream>>>(
            B2p[i], dims[i + 1], Bf + boff[i][1], padd[i + 1]);
    }

    // 5 GIN layers
    const float* h = x;
    for (int i = 0; i < 5; ++i) {
        int Kin = padd[i];
        int Dp  = padd[i + 1];
        int Da  = dims[i + 1];

        long long cn = (long long)NN * Kin / 4;
        copy4_kernel<<<(unsigned)((cn + 255) / 256), 256, 0, stream>>>(Mb, h, cn);

        long long sn = (long long)EE * (Kin / 4);
        scatter_add_kernel<<<(unsigned)((sn + 255) / 256), 256, 0, stream>>>(
            ei, h, Mb, EE, Kin);

        dim3 grid(Dp / 64, (NN + TM - 1) / TM);
        wmma_gemm_bias_act<<<grid, 256, 0, stream>>>(
            Mb, Wbf + woff[i][0], Bf + boff[i][0], T, NN, Kin, Dp, Da, 1);
        wmma_gemm_bias_act<<<grid, 256, 0, stream>>>(
            T, Wbf + woff[i][1], Bf + boff[i][1], H, NN, Dp, Dp, Da, (i < 4) ? 1 : 0);
        h = H;
    }

    // Global mean pool + LayerNorm
    long long zn = 256 * 768 + 256;
    zero_kernel<<<(unsigned)((zn + 255) / 256), 256, 0, stream>>>(sums, zn);
    long long pn = (long long)NN * (768 / 4);
    pool_sum_kernel<<<(unsigned)((pn + 255) / 256), 256, 0, stream>>>(H, batch, sums, NN);
    count_kernel<<<(NN + 255) / 256, 256, 0, stream>>>(batch, counts, NN);
    ln_kernel<<<256, 256, 0, stream>>>(sums, counts, lnw, lnb, (float*)d_out);
}
